// TEDClassifier_70592082477721
// MI455X (gfx1250) — matrix operand order repaired and checked by reading, not compiled, not run on hardware
//
#include <hip/hip_runtime.h>
#include <cstdint>
#include <cstddef>

// ---------------------------------------------------------------------------
// Types for CDNA5 WMMA (wave32)
// ---------------------------------------------------------------------------
typedef __bf16 bf16;
typedef __attribute__((ext_vector_type(16))) __bf16 bf16x16;
typedef __attribute__((ext_vector_type(8)))  float   floatx8;
typedef __attribute__((ext_vector_type(4)))  unsigned int u32x4;

// ---------------------------------------------------------------------------
// Small helpers
// ---------------------------------------------------------------------------
__device__ inline void atomicMaxF(float* addr, float v) {
  // Ordered-int trick: works across mixed signs.
  if (v >= 0.0f) atomicMax((int*)addr, __float_as_int(v));
  else           atomicMin((unsigned int*)addr, __float_as_uint(v));
}

__global__ void fill_f32(float* p, float v, long long n) {
  long long i = (long long)blockIdx.x * blockDim.x + threadIdx.x;
  if (i < n) p[i] = v;
}

__global__ void convert_f32_bf16(const float* __restrict__ src, bf16* __restrict__ dst,
                                 long long n) {
  long long i = (long long)blockIdx.x * blockDim.x + threadIdx.x;
  if (i < n) dst[i] = (bf16)src[i];
}

// transpose + convert: dst[n*K + k] = (bf16) src[k*N + n]   (W is [K][N] row-major)
__global__ void transpose_w_bf16(const float* __restrict__ src, bf16* __restrict__ dst,
                                 int K, int N) {
  long long i = (long long)blockIdx.x * blockDim.x + threadIdx.x;
  if (i >= (long long)K * N) return;
  int k = (int)(i / N), n = (int)(i % N);
  dst[(size_t)n * K + k] = (bf16)src[i];
}

// concat up to 3 bias segments (null => zeros)
__global__ void concat_bias(float* dst, const float* s0, const float* s1, const float* s2,
                            int n0, int n1, int n2) {
  int i = blockIdx.x * blockDim.x + threadIdx.x;
  int tot = n0 + n1 + n2;
  if (i >= tot) return;
  float v;
  if (i < n0)            v = s0 ? s0[i] : 0.0f;
  else if (i < n0 + n1)  v = s1 ? s1[i - n0] : 0.0f;
  else                   v = s2 ? s2[i - n0 - n1] : 0.0f;
  dst[i] = v;
}

// ---------------------------------------------------------------------------
// role_tbl[r][d] = sum_e role_emb[r][e] * role_proj_W[e][d]   (7 x 64)
// ---------------------------------------------------------------------------
__global__ void role_table_kernel(const float* __restrict__ emb,
                                  const float* __restrict__ W,
                                  float* __restrict__ tbl) {
  int r = blockIdx.x, d = threadIdx.x;       // 7 blocks x 64 threads
  float acc = 0.0f;
  for (int e = 0; e < 64; ++e) acc += emb[r * 64 + e] * W[e * 64 + d];
  tbl[r * 64 + d] = acc;
}

// nf_bf[i][j] = j<768 ? text[i][j] : role_tbl[role_ids[i]][j-768]
__global__ void build_nf_bf16(const float* __restrict__ text, const int* __restrict__ role_ids,
                              const float* __restrict__ tbl, bf16* __restrict__ nf,
                              long long n_elems) {
  long long idx = (long long)blockIdx.x * blockDim.x + threadIdx.x;
  if (idx >= n_elems) return;
  int i = (int)(idx / 832), j = (int)(idx % 832);
  float v = (j < 768) ? text[(size_t)i * 768 + j] : tbl[role_ids[i] * 64 + (j - 768)];
  nf[idx] = (bf16)v;
}

// ---------------------------------------------------------------------------
// WMMA GEMM: C[M][Ncols] = A[M][K](bf16) * Wt[Ncols][K](bf16)^T (+bias[col])
// One wave computes a 32x64 tile: 8 x v_wmma_f32_16x16x32_bf16 per 32-wide k-step,
// double-buffered so next k-step's global loads are in flight during the MMAs.
// __launch_bounds__(256,1) => allow full VGPR allocation (no spills).
// Requires: M%32==0, Ncols%64==0, K%64==0 (even number of 32-k-steps),
// 16B row alignment.
// ---------------------------------------------------------------------------
__device__ inline bf16x16 ld_frag(const bf16* p0, const bf16* p1) {
  union { bf16x16 v; u32x4 q[2]; } u;
  u.q[0] = *(const u32x4*)p0;
  u.q[1] = *(const u32x4*)p1;
  return u.v;
}

__global__ void __launch_bounds__(256, 1)
gemm_wmma(const bf16* __restrict__ A, int lda,
          const bf16* __restrict__ Wt, int ldw,
          const float* __restrict__ bias,
          float* __restrict__ C, int ldc,
          int M, int Ncols, int K) {
  int gid  = blockIdx.x * blockDim.x + threadIdx.x;
  int wave = gid >> 5;
  int lane = threadIdx.x & 31;
  int ntn  = Ncols >> 6;
  int tiles = (M >> 5) * ntn;
  if (wave >= tiles) return;
  int mt = wave / ntn, nt = wave % ntn;
  int m0 = mt << 5, n0 = nt << 6;
  int hs  = lane >> 4;     // half select (0/1)
  int l15 = lane & 15;

  // A fragment base pointers for the two 16-row sub-tiles.
  // ISA 16-bit A layout: lanes 0-15 hold K {0-7,16-23}, lanes 16-31 hold K {8-15,24-31}
  const bf16* arow0 = A + (size_t)(m0 + l15) * lda + hs * 8;
  const bf16* arow1 = arow0 + (size_t)16 * lda;
  // B fragment base: lane = column, lanes 0-15: K {0-7,16-23}, lanes 16-31: K {8-15,24-31}
  const bf16* brow = Wt + (size_t)(n0 + l15) * ldw + hs * 8;
  const size_t bstep = (size_t)16 * ldw;

  floatx8 acc[2][4];
#pragma unroll
  for (int m = 0; m < 2; ++m)
#pragma unroll
    for (int s = 0; s < 4; ++s) acc[m][s] = (floatx8){0,0,0,0,0,0,0,0};

  bf16x16 a0[2], a1[2], b0[4], b1[4];
  // prologue: buffer 0 <- k = 0
  a0[0] = ld_frag(arow0, arow0 + 16);
  a0[1] = ld_frag(arow1, arow1 + 16);
#pragma unroll
  for (int s = 0; s < 4; ++s) {
    const bf16* p = brow + s * bstep;
    b0[s] = ld_frag(p, p + 16);
  }

  for (int k0 = 0; k0 < K; k0 += 64) {
    // issue loads for k0+32 (buffer 1) before consuming buffer 0
    {
      const bf16* p = arow0 + k0 + 32; a1[0] = ld_frag(p, p + 16);
      p = arow1 + k0 + 32;             a1[1] = ld_frag(p, p + 16);
#pragma unroll
      for (int s = 0; s < 4; ++s) {
        const bf16* q = brow + s * bstep + k0 + 32;
        b1[s] = ld_frag(q, q + 16);
      }
    }
#pragma unroll
    for (int s = 0; s < 4; ++s) {
      acc[0][s] = __builtin_amdgcn_wmma_f32_16x16x32_bf16(
          false, a0[0], false, b0[s], (short)0, acc[0][s], false, false);
      acc[1][s] = __builtin_amdgcn_wmma_f32_16x16x32_bf16(
          false, a0[1], false, b0[s], (short)0, acc[1][s], false, false);
    }
    // issue loads for k0+64 (buffer 0) before consuming buffer 1
    if (k0 + 64 < K) {
      const bf16* p = arow0 + k0 + 64; a0[0] = ld_frag(p, p + 16);
      p = arow1 + k0 + 64;             a0[1] = ld_frag(p, p + 16);
#pragma unroll
      for (int s = 0; s < 4; ++s) {
        const bf16* q = brow + s * bstep + k0 + 64;
        b0[s] = ld_frag(q, q + 16);
      }
    }
#pragma unroll
    for (int s = 0; s < 4; ++s) {
      acc[0][s] = __builtin_amdgcn_wmma_f32_16x16x32_bf16(
          false, a1[0], false, b1[s], (short)0, acc[0][s], false, false);
      acc[1][s] = __builtin_amdgcn_wmma_f32_16x16x32_bf16(
          false, a1[1], false, b1[s], (short)0, acc[1][s], false, false);
    }
  }

  // C/D layout: VGPR r -> row base + hs*8 + r, column = n0 + s*16 + l15
#pragma unroll
  for (int mm = 0; mm < 2; ++mm) {
    int rbase = m0 + mm * 16 + hs * 8;
#pragma unroll
    for (int s = 0; s < 4; ++s) {
      int col = n0 + s * 16 + l15;
      float bv = bias ? bias[col] : 0.0f;
#pragma unroll
      for (int r = 0; r < 8; ++r)
        C[(size_t)(rbase + r) * ldc + col] = acc[mm][s][r] + bv;
    }
  }
}

// ---------------------------------------------------------------------------
// GATv2 edge pass 1: scores + per-target max (atomic).
// a[e][h] = sum_d leaky_relu(xl[src][d] + xr[tgt][d], 0.2) * att[h*64 + (d%64)]
// ---------------------------------------------------------------------------
__global__ void edge_scores(const int* __restrict__ src_idx, const int* __restrict__ dst_idx,
                            int E, int Etot,
                            const float* __restrict__ xl, const float* __restrict__ xr, int ld,
                            const float* __restrict__ att,
                            float* __restrict__ a_e, float* __restrict__ amax) {
  __shared__ float satt[256];
  satt[threadIdx.x] = att[threadIdx.x];
  __syncthreads();
  int e = blockIdx.x * 256 + threadIdx.x;
  if (e >= Etot) return;
  int s = (e < E) ? src_idx[e] : (e - E);
  int t = (e < E) ? dst_idx[e] : (e - E);
  const float4* pl = (const float4*)(xl + (size_t)s * ld);
  const float4* pr = (const float4*)(xr + (size_t)t * ld);
  float acc[4] = {0.f, 0.f, 0.f, 0.f};
  for (int q = 0; q < 64; ++q) {
    float4 l = pl[q], r = pr[q];
    int h = q >> 4;
    float m;
    m = l.x + r.x; m = m > 0.f ? m : 0.2f * m; acc[h] += m * satt[q * 4 + 0];
    m = l.y + r.y; m = m > 0.f ? m : 0.2f * m; acc[h] += m * satt[q * 4 + 1];
    m = l.z + r.z; m = m > 0.f ? m : 0.2f * m; acc[h] += m * satt[q * 4 + 2];
    m = l.w + r.w; m = m > 0.f ? m : 0.2f * m; acc[h] += m * satt[q * 4 + 3];
  }
#pragma unroll
  for (int h = 0; h < 4; ++h) {
    a_e[(size_t)e * 4 + h] = acc[h];
    atomicMaxF(&amax[(size_t)t * 4 + h], acc[h]);
  }
}

// pass 2: ea = exp(a - amax[tgt]); den[tgt][h] += ea; gout[tgt][:] += xl[src][:]*ea
// One wave per edge; lane covers 8 contiguous dims (head = lane/8).
__global__ void edge_aggregate(const int* __restrict__ src_idx, const int* __restrict__ dst_idx,
                               int E, int Etot,
                               const float* __restrict__ xl, int ld,
                               const float* __restrict__ a_e, const float* __restrict__ amax,
                               float* __restrict__ den, float* __restrict__ gout) {
  int tid = threadIdx.x, lane = tid & 31;
  int e = blockIdx.x * 8 + (tid >> 5);
  if (e >= Etot) return;
  int s = (e < E) ? src_idx[e] : (e - E);
  int t = (e < E) ? dst_idx[e] : (e - E);
  int h = lane >> 3;
  float ea = __expf(a_e[(size_t)e * 4 + h] - amax[(size_t)t * 4 + h]);
  if ((lane & 7) == 0) atomicAdd(&den[(size_t)t * 4 + h], ea);
  const float4* ps = (const float4*)(xl + (size_t)s * ld + lane * 8);
  float4 v0 = ps[0], v1 = ps[1];
  float* po = gout + (size_t)t * 256 + lane * 8;
  atomicAdd(po + 0, v0.x * ea); atomicAdd(po + 1, v0.y * ea);
  atomicAdd(po + 2, v0.z * ea); atomicAdd(po + 3, v0.w * ea);
  atomicAdd(po + 4, v1.x * ea); atomicAdd(po + 5, v1.y * ea);
  atomicAdd(po + 6, v1.z * ea); atomicAdd(po + 7, v1.w * ea);
}

// Layer-1 finalize: t = gout/den + g1_bias; LN(256); + residual; ELU; -> bf16
__global__ void finalize_gat1(const float* __restrict__ gout, const float* __restrict__ den,
                              const float* __restrict__ g1_bias,
                              const float* __restrict__ ln_g, const float* __restrict__ ln_b,
                              const float* __restrict__ xres, int ldres,
                              bf16* __restrict__ x1bf) {
  int n = blockIdx.x, d = threadIdx.x;           // 256 threads
  __shared__ float sh[256];
  float t = gout[(size_t)n * 256 + d] / fmaxf(den[(size_t)n * 4 + (d >> 6)], 1e-16f)
            + g1_bias[d];
  sh[d] = t; __syncthreads();
  for (int off = 128; off > 0; off >>= 1) { if (d < off) sh[d] += sh[d + off]; __syncthreads(); }
  float mean = sh[0] * (1.0f / 256.0f); __syncthreads();
  float dv = t - mean;
  sh[d] = dv * dv; __syncthreads();
  for (int off = 128; off > 0; off >>= 1) { if (d < off) sh[d] += sh[d + off]; __syncthreads(); }
  float var = sh[0] * (1.0f / 256.0f);
  float y = dv * rsqrtf(var + 1e-5f) * ln_g[d] + ln_b[d] + xres[(size_t)n * ldres + d];
  float x = y > 0.f ? y : (__expf(y) - 1.0f);    // ELU
  x1bf[(size_t)n * 256 + d] = (bf16)x;
}

// Layer-2 finalize: mean over heads + g2_bias; LN(64); -> bf16
__global__ void finalize_gat2(const float* __restrict__ gout, const float* __restrict__ den,
                              const float* __restrict__ g2_bias,
                              const float* __restrict__ ln_g, const float* __restrict__ ln_b,
                              bf16* __restrict__ x2bf) {
  int n = blockIdx.x, d = threadIdx.x;           // 64 threads
  __shared__ float sh[64];
  float t = 0.f;
#pragma unroll
  for (int h = 0; h < 4; ++h)
    t += gout[(size_t)n * 256 + h * 64 + d] / fmaxf(den[(size_t)n * 4 + h], 1e-16f);
  t = t * 0.25f + g2_bias[d];
  sh[d] = t; __syncthreads();
  for (int off = 32; off > 0; off >>= 1) { if (d < off) sh[d] += sh[d + off]; __syncthreads(); }
  float mean = sh[0] * (1.0f / 64.0f); __syncthreads();
  float dv = t - mean;
  sh[d] = dv * dv; __syncthreads();
  for (int off = 32; off > 0; off >>= 1) { if (d < off) sh[d] += sh[d + off]; __syncthreads(); }
  float var = sh[0] * (1.0f / 64.0f);
  float y = dv * rsqrtf(var + 1e-5f) * ln_g[d] + ln_b[d];
  x2bf[(size_t)n * 64 + d] = (bf16)y;
}

// ---------------------------------------------------------------------------
// Segment bookkeeping for to_dense_batch semantics
// ---------------------------------------------------------------------------
__global__ void count_kernel(const int* __restrict__ batch, int n, int* counts) {
  int i = blockIdx.x * blockDim.x + threadIdx.x;
  if (i < n) atomicAdd(&counts[batch[i]], 1);
}
__global__ void starts_scan(const int* __restrict__ counts, int* starts, int B) {
  int b = threadIdx.x;
  if (b >= B) return;
  int s = 0;
  for (int j = 0; j < b; ++j) s += counts[j];
  starts[b] = s;
}

// ---------------------------------------------------------------------------
// Small dense GEMM for B=32 rows: out[r][c] = bias[c] + sum_k A[r][k]*W[k][c] (+resid)
// grid = rows, blockDim = Ncol
// ---------------------------------------------------------------------------
__global__ void small_gemm(const float* __restrict__ A, int lda,
                           const float* __restrict__ W,
                           const float* __restrict__ bias,
                           const float* __restrict__ resid, int ldr,
                           float* __restrict__ out, int ldo, int K, int Ncol) {
  int r = blockIdx.x, c = threadIdx.x;
  float acc = bias ? bias[c] : 0.0f;
  const float* a = A + (size_t)r * lda;
  for (int k = 0; k < K; ++k) acc += a[k] * W[(size_t)k * Ncol + c];
  if (resid) acc += resid[(size_t)r * ldr + c];
  out[(size_t)r * ldo + c] = acc;
}

// Row LayerNorm (blockDim = D, power of two), optional exact GELU
__global__ void ln_rows(const float* __restrict__ x, int ldx,
                        const float* __restrict__ g, const float* __restrict__ b,
                        float* __restrict__ out, int ldo, int gelu) {
  extern __shared__ float sh[];
  int r = blockIdx.x, d = threadIdx.x, D = blockDim.x;
  float t = x[(size_t)r * ldx + d];
  sh[d] = t; __syncthreads();
  for (int off = D >> 1; off > 0; off >>= 1) { if (d < off) sh[d] += sh[d + off]; __syncthreads(); }
  float mean = sh[0] / (float)D; __syncthreads();
  float dv = t - mean;
  sh[d] = dv * dv; __syncthreads();
  for (int off = D >> 1; off > 0; off >>= 1) { if (d < off) sh[d] += sh[d + off]; __syncthreads(); }
  float var = sh[0] / (float)D;
  float y = dv * rsqrtf(var + 1e-5f) * g[d] + b[d];
  if (gelu) y = 0.5f * y * (1.0f + erff(y * 0.70710678118654752f));
  out[(size_t)r * ldo + d] = y;
}

// ---------------------------------------------------------------------------
// Cross attention: 1 query per graph, 4 heads of 64. Block per (b,h), 64 threads.
// ---------------------------------------------------------------------------
__global__ void cross_attn(const float* __restrict__ q, const float* __restrict__ kbuf,
                           const float* __restrict__ vbuf,
                           const int* __restrict__ starts, const int* __restrict__ counts,
                           float* __restrict__ o) {
  int b = blockIdx.x >> 2, h = blockIdx.x & 3;
  int tid = threadIdx.x;                                 // 64 threads
  __shared__ float sc[1024];
  __shared__ float sred[64];
  __shared__ float qsh[64];
  int st = starts[b], cnt = counts[b];
  if (cnt > 1024) cnt = 1024;                            // NMAX clamp (to_dense_batch)
  qsh[tid] = q[(size_t)b * 256 + h * 64 + tid];
  __syncthreads();
  float lmax = -1e30f;
  for (int i = tid; i < cnt; i += 64) {
    const float* kp = kbuf + (size_t)(st + i) * 256 + h * 64;
    float sdot = 0.f;
    for (int d = 0; d < 64; ++d) sdot += qsh[d] * kp[d];
    sdot *= 0.125f;                                      // 1/sqrt(64)
    sc[i] = sdot;
    lmax = fmaxf(lmax, sdot);
  }
  sred[tid] = lmax; __syncthreads();
  for (int off = 32; off > 0; off >>= 1) { if (tid < off) sred[tid] = fmaxf(sred[tid], sred[tid + off]); __syncthreads(); }
  float mx = sred[0]; __syncthreads();
  float lsum = 0.f;
  for (int i = tid; i < cnt; i += 64) { float ea = __expf(sc[i] - mx); sc[i] = ea; lsum += ea; }
  sred[tid] = lsum; __syncthreads();
  for (int off = 32; off > 0; off >>= 1) { if (tid < off) sred[tid] += sred[tid + off]; __syncthreads(); }
  float dsum = sred[0];
  float inv = dsum > 0.f ? 1.0f / dsum : 0.0f;
  float acc = 0.f;
  for (int i = 0; i < cnt; ++i) acc += sc[i] * vbuf[(size_t)(st + i) * 256 + h * 64 + tid];
  o[(size_t)b * 256 + h * 64 + tid] = acc * inv;
}

// ---------------------------------------------------------------------------
// Masked mean pools: global mean -> combined[:,0:256], |pro - opp| -> [:,512:768]
// ---------------------------------------------------------------------------
__global__ void pools(const float* __restrict__ nprj, const int* __restrict__ role_ids,
                      const int* __restrict__ starts, const int* __restrict__ counts,
                      float* __restrict__ combined) {
  int b = blockIdx.x, d = threadIdx.x;                   // 256 threads
  int st = starts[b], cnt = counts[b];
  if (cnt > 1024) cnt = 1024;
  float gs = 0.f, ps = 0.f, os = 0.f;
  int cp = 0, co = 0;
  for (int i = 0; i < cnt; ++i) {
    int n = st + i;
    float v = nprj[(size_t)n * 256 + d];
    gs += v;
    int r = role_ids[n];
    if (r == 0 || r == 2 || r == 4)      { ps += v; cp++; }
    else if (r == 1 || r == 3 || r == 5) { os += v; co++; }
  }
  combined[(size_t)b * 768 + d]       = gs / fmaxf((float)cnt, 1.0f);
  combined[(size_t)b * 768 + 512 + d] = fabsf(ps / fmaxf((float)cp, 1.0f)
                                            - os / fmaxf((float)co, 1.0f));
}

// ---------------------------------------------------------------------------
// Launcher
// ---------------------------------------------------------------------------
extern "C" void kernel_launch(void* const* d_in, const int* in_sizes, int n_in,
                              void* d_out, int out_size, void* d_ws, size_t ws_size,
                              hipStream_t stream) {
  // ---- inputs (setup_inputs dict order, params flattened in insertion order) ----
  const float* node_text  = (const float*)d_in[0];   // [N,768]
  const float* news       = (const float*)d_in[1];   // [32,768]
  const int*   role_ids   = (const int*)  d_in[2];   // [N]
  const int*   eidx       = (const int*)  d_in[3];   // [2,E]
  const int*   batch      = (const int*)  d_in[4];   // [N]
  const float* p_role_emb = (const float*)d_in[5];
  const float* p_role_W   = (const float*)d_in[6];
  const float* p_gat_in_W = (const float*)d_in[7];
  const float* p_g1_Wl    = (const float*)d_in[8];
  const float* p_g1_bl    = (const float*)d_in[9];
  const float* p_g1_Wr    = (const float*)d_in[10];
  const float* p_g1_br    = (const float*)d_in[11];
  const float* p_g1_att   = (const float*)d_in[12];
  const float* p_g1_bias  = (const float*)d_in[13];
  const float* p_ln1_g    = (const float*)d_in[14];
  const float* p_ln1_b    = (const float*)d_in[15];
  const float* p_g2_Wl    = (const float*)d_in[16];
  const float* p_g2_bl    = (const float*)d_in[17];
  const float* p_g2_Wr    = (const float*)d_in[18];
  const float* p_g2_br    = (const float*)d_in[19];
  const float* p_g2_att   = (const float*)d_in[20];
  const float* p_g2_bias  = (const float*)d_in[21];
  const float* p_ln2_g    = (const float*)d_in[22];
  const float* p_ln2_b    = (const float*)d_in[23];
  const float* p_np_W     = (const float*)d_in[24];
  const float* p_np_b     = (const float*)d_in[25];
  const float* p_news_W   = (const float*)d_in[26];
  const float* p_news_b   = (const float*)d_in[27];
  const float* p_Wq = (const float*)d_in[28]; const float* p_bq = (const float*)d_in[29];
  const float* p_Wk = (const float*)d_in[30]; const float* p_bk = (const float*)d_in[31];
  const float* p_Wv = (const float*)d_in[32]; const float* p_bv = (const float*)d_in[33];
  const float* p_Wo = (const float*)d_in[34]; const float* p_bo = (const float*)d_in[35];
  const float* p_can_g = (const float*)d_in[36]; const float* p_can_b = (const float*)d_in[37];
  const float* p_cls_W1 = (const float*)d_in[38]; const float* p_cls_b1 = (const float*)d_in[39];
  const float* p_cls_lng = (const float*)d_in[40]; const float* p_cls_lnb = (const float*)d_in[41];
  const float* p_cls_W2 = (const float*)d_in[42]; const float* p_cls_b2 = (const float*)d_in[43];

  const int N = in_sizes[2];          // 20000
  const int E = in_sizes[3] / 2;      // 320000
  const int Etot = E + N;             // with self loops
  const int B = 32;
  const int* src_idx = eidx;
  const int* dst_idx = eidx + E;

  // ---- workspace carving (256B aligned), with region reuse ----
  char* ws = (char*)d_ws;
  size_t off = 0;
  auto carve = [&](size_t bytes) -> void* {
    off = (off + 255) & ~(size_t)255;
    void* p = ws + off;
    off += bytes;
    return p;
  };
  bf16* wt_big   = (bf16*)carve((size_t)768 * 832 * 2);   // [gat_in|g1_Wl|g1_Wr]^T
  bf16* wt2      = (bf16*)carve((size_t)512 * 256 * 2);   // [g2_Wl|g2_Wr]^T
  bf16* wtnp     = (bf16*)carve((size_t)256 * 64 * 2);
  bf16* wtk      = (bf16*)carve((size_t)256 * 256 * 2);
  bf16* wtv      = (bf16*)carve((size_t)256 * 256 * 2);
  float* bias768 = (float*)carve(768 * 4);
  float* bias512 = (float*)carve(512 * 4);
  float* role_tbl= (float*)carve(7 * 64 * 4);
  void* R2 = carve((size_t)N * 832 * 2);                  // nf_bf -> kbuf
  void* R1 = carve((size_t)N * 768 * 4);                  // C0 -> C2 -> vbuf
  float* a_e  = (float*)carve((size_t)Etot * 4 * 4);
  float* amax = (float*)carve((size_t)N * 4 * 4);
  float* den  = (float*)carve((size_t)N * 4 * 4);
  float* gout = (float*)carve((size_t)N * 256 * 4);       // gat accum -> node_proj
  bf16*  x1bf = (bf16*)carve((size_t)N * 256 * 2);        // x1 -> np_bf
  bf16*  x2bf = (bf16*)carve((size_t)N * 64 * 2);
  int*   counts = (int*)carve(B * 4);
  int*   starts = (int*)carve(B * 4);
  float* e_proj = (float*)carve(B * 256 * 4);
  float* qbuf   = (float*)carve(B * 256 * 4);
  float* obuf   = (float*)carve(B * 256 * 4);
  float* tmp32  = (float*)carve(B * 256 * 4);
  float* combined = (float*)carve(B * 768 * 4);
  float* h1 = (float*)carve(B * 256 * 4);
  float* h2 = (float*)carve(B * 256 * 4);

  bf16*  nf_bf = (bf16*)R2;   float* kbuf = (float*)R2;
  float* C0 = (float*)R1;     float* C2 = (float*)R1;     float* vbuf = (float*)R1;
  float* nprj = gout;         bf16* npbf = x1bf;

  auto cdiv = [](long long a, long long b) { return (int)((a + b - 1) / b); };

  // ---- 1) tiny precompute: role table, weights -> bf16 transposed, bias concat ----
  role_table_kernel<<<7, 64, 0, stream>>>(p_role_emb, p_role_W, role_tbl);
  transpose_w_bf16<<<cdiv((long long)832 * 256, 256), 256, 0, stream>>>(p_gat_in_W, wt_big, 832, 256);
  transpose_w_bf16<<<cdiv((long long)832 * 256, 256), 256, 0, stream>>>(p_g1_Wl, wt_big + (size_t)256 * 832, 832, 256);
  transpose_w_bf16<<<cdiv((long long)832 * 256, 256), 256, 0, stream>>>(p_g1_Wr, wt_big + (size_t)512 * 832, 832, 256);
  transpose_w_bf16<<<cdiv((long long)256 * 256, 256), 256, 0, stream>>>(p_g2_Wl, wt2, 256, 256);
  transpose_w_bf16<<<cdiv((long long)256 * 256, 256), 256, 0, stream>>>(p_g2_Wr, wt2 + (size_t)256 * 256, 256, 256);
  transpose_w_bf16<<<cdiv((long long)64 * 256, 256), 256, 0, stream>>>(p_np_W, wtnp, 64, 256);
  transpose_w_bf16<<<cdiv((long long)256 * 256, 256), 256, 0, stream>>>(p_Wk, wtk, 256, 256);
  transpose_w_bf16<<<cdiv((long long)256 * 256, 256), 256, 0, stream>>>(p_Wv, wtv, 256, 256);
  concat_bias<<<3, 256, 0, stream>>>(bias768, nullptr, p_g1_bl, p_g1_br, 256, 256, 256);
  concat_bias<<<2, 256, 0, stream>>>(bias512, p_g2_bl, p_g2_br, nullptr, 256, 256, 0);

  // ---- 2) node features -> bf16 ([text | role_proj]) ----
  build_nf_bf16<<<cdiv((long long)N * 832, 256), 256, 0, stream>>>(node_text, role_ids, role_tbl, nf_bf, (long long)N * 832);

  // ---- 3) fused GEMM: C0[N,768] = nf @ [gat_in|Wl|Wr] (+0|bl|br) ----
  {
    int tiles = (N / 32) * (768 / 64);
    gemm_wmma<<<cdiv(tiles, 8), 256, 0, stream>>>(nf_bf, 832, wt_big, 832, bias768, C0, 768, N, 768, 832);
  }

  // ---- 4) GAT layer 1 (edges incl. self loops) ----
  fill_f32<<<cdiv((long long)N * 4, 256), 256, 0, stream>>>(amax, -1e30f, (long long)N * 4);
  fill_f32<<<cdiv((long long)N * 4, 256), 256, 0, stream>>>(den, 0.0f, (long long)N * 4);
  fill_f32<<<cdiv((long long)N * 256, 256), 256, 0, stream>>>(gout, 0.0f, (long long)N * 256);
  edge_scores<<<cdiv(Etot, 256), 256, 0, stream>>>(src_idx, dst_idx, E, Etot, C0 + 256, C0 + 512, 768, p_g1_att, a_e, amax);
  edge_aggregate<<<cdiv(Etot, 8), 256, 0, stream>>>(src_idx, dst_idx, E, Etot, C0 + 256, 768, a_e, amax, den, gout);
  finalize_gat1<<<N, 256, 0, stream>>>(gout, den, p_g1_bias, p_ln1_g, p_ln1_b, C0, 768, x1bf);

  // ---- 5) GAT layer 2 ----
  {
    int tiles = (N / 32) * (512 / 64);
    gemm_wmma<<<cdiv(tiles, 8), 256, 0, stream>>>(x1bf, 256, wt2, 256, bias512, C2, 512, N, 512, 256);
  }
  fill_f32<<<cdiv((long long)N * 4, 256), 256, 0, stream>>>(amax, -1e30f, (long long)N * 4);
  fill_f32<<<cdiv((long long)N * 4, 256), 256, 0, stream>>>(den, 0.0f, (long long)N * 4);
  fill_f32<<<cdiv((long long)N * 256, 256), 256, 0, stream>>>(gout, 0.0f, (long long)N * 256);
  edge_scores<<<cdiv(Etot, 256), 256, 0, stream>>>(src_idx, dst_idx, E, Etot, C2, C2 + 256, 512, p_g2_att, a_e, amax);
  edge_aggregate<<<cdiv(Etot, 8), 256, 0, stream>>>(src_idx, dst_idx, E, Etot, C2, 512, a_e, amax, den, gout);
  finalize_gat2<<<N, 64, 0, stream>>>(gout, den, p_g2_bias, p_ln2_g, p_ln2_b, x2bf);

  // ---- 6) node projection + K/V for cross attention (all via WMMA) ----
  {
    int tiles = (N / 32) * (256 / 64);
    gemm_wmma<<<cdiv(tiles, 8), 256, 0, stream>>>(x2bf, 64, wtnp, 64, p_np_b, nprj, 256, N, 256, 64);
    convert_f32_bf16<<<cdiv((long long)N * 256, 256), 256, 0, stream>>>(nprj, npbf, (long long)N * 256);
    gemm_wmma<<<cdiv(tiles, 8), 256, 0, stream>>>(npbf, 256, wtk, 256, p_bk, kbuf, 256, N, 256, 256);
    gemm_wmma<<<cdiv(tiles, 8), 256, 0, stream>>>(npbf, 256, wtv, 256, p_bv, vbuf, 256, N, 256, 256);
  }

  // ---- 7) per-graph segments (batch is sorted) ----
  fill_f32<<<1, 64, 0, stream>>>((float*)counts, 0.0f, B);
  count_kernel<<<cdiv(N, 256), 256, 0, stream>>>(batch, N, counts);
  starts_scan<<<1, 32, 0, stream>>>(counts, starts, B);

  // ---- 8) news proj, query, cross attention, Wo + residual + LN ----
  small_gemm<<<B, 256, 0, stream>>>(news, 768, p_news_W, p_news_b, nullptr, 0, e_proj, 256, 768, 256);
  small_gemm<<<B, 256, 0, stream>>>(e_proj, 256, p_Wq, p_bq, nullptr, 0, qbuf, 256, 256, 256);
  cross_attn<<<B * 4, 64, 0, stream>>>(qbuf, kbuf, vbuf, starts, counts, obuf);
  small_gemm<<<B, 256, 0, stream>>>(obuf, 256, p_Wo, p_bo, e_proj, 256, tmp32, 256, 256, 256);
  ln_rows<<<B, 256, 256 * 4, stream>>>(tmp32, 256, p_can_g, p_can_b, combined + 256, 768, 0);

  // ---- 9) pools + classifier ----
  pools<<<B, 256, 0, stream>>>(nprj, role_ids, starts, counts, combined);
  small_gemm<<<B, 256, 0, stream>>>(combined, 768, p_cls_W1, p_cls_b1, nullptr, 0, h1, 256, 768, 256);
  ln_rows<<<B, 256, 256 * 4, stream>>>(h1, 256, p_cls_lng, p_cls_lnb, h2, 256, 1);
  small_gemm<<<B, 2, 0, stream>>>(h2, 256, p_cls_W2, p_cls_b2, nullptr, 0, (float*)d_out, 2, 256, 2);
}
